// MultiheadAttention_42726334660933
// MI455X (gfx1250) — matrix operand order repaired and checked by reading, not compiled, not run on hardware
//
#include <hip/hip_runtime.h>
#include <hip/hip_bf16.h>
#include <stdint.h>

#define D_EMBED 1024
#define N_HEADS 16
#define HD      64
#define DFF_    4096
#define BATCH   2
#define SEQ     2048
#define NROWS   (BATCH * SEQ)   // 4096

// GEMM block tiling
#define BM  128
#define BN  128
#define BK  32
#define BKP 40      // padded LDS row stride (32 elems + 8 pad) -> 80B, breaks bank pattern

typedef __attribute__((ext_vector_type(16))) __bf16 bf16x16;
typedef __attribute__((ext_vector_type(8)))  float  f32x8;
typedef __attribute__((ext_vector_type(4)))  unsigned int u32x4;
typedef __attribute__((ext_vector_type(8)))  int  i32x8;
typedef __attribute__((ext_vector_type(4)))  int  i32x4;

union FragAB { bf16x16 v; uint4 u[2]; };

__device__ __forceinline__ __bf16 f2bf(float f) {
    union { float f; uint32_t u; } a; a.f = f;
    uint32_t u = a.u;
    uint32_t r = (u + 0x7FFFu + ((u >> 16) & 1u)) >> 16;   // RTNE
    union { unsigned short s; __bf16 b; } o; o.s = (unsigned short)r;
    return o.b;
}

__device__ __forceinline__ f32x8 wmma_bf16(const FragAB& a, const FragAB& b, f32x8 c) {
    return __builtin_amdgcn_wmma_f32_16x16x32_bf16(false, a.v, false, b.v,
                                                   (short)0, c, false, false);
}

// A-matrix 16x32 fragment (ISA 7.12.2): lane m=l&15; K runs [k0+g*8,+8) and [k0+16+g*8,+8)
__device__ __forceinline__ FragAB load_fragA(const __bf16* rowPtr, int k0, int g) {
    FragAB f;
    f.u[0] = *reinterpret_cast<const uint4*>(rowPtr + k0 + g * 8);
    f.u[1] = *reinterpret_cast<const uint4*>(rowPtr + k0 + 16 + g * 8);
    return f;
}
// B-matrix 32x16 fragment: lane n=l&15; K runs [k0+g*8,+8) and [k0+16+g*8,+8)
__device__ __forceinline__ FragAB load_fragB(const __bf16* colPtr, int k0, int g) {
    FragAB f;
    f.u[0] = *reinterpret_cast<const uint4*>(colPtr + k0 + g * 8);
    f.u[1] = *reinterpret_cast<const uint4*>(colPtr + k0 + 16 + g * 8);
    return f;
}

// ------------------------------------------------------- TDM 2D tile -> LDS
#if __has_builtin(__builtin_amdgcn_tensor_load_to_lds)
#define HAVE_TDM 1
// Load a [tile_d1 rows x tile_d0 elems] bf16 tile (row stride = stride_d0 elems
// in memory) into LDS at lds_byte_off with 4-DWORD padding per 16-DWORD row
// (=> LDS row stride BKP elems). D# packing per CDNA5 ISA §8.3/8.4.
__device__ __forceinline__ void tdm_load_2d(uint32_t lds_byte_off,
                                            const __bf16* gptr,
                                            uint32_t tile_d0, uint32_t tile_d1,
                                            uint32_t tensor_d0, uint32_t tensor_d1,
                                            uint64_t stride_d0_elems) {
    uint64_t ga = (uint64_t)(uintptr_t)gptr;
    u32x4 g0;
    g0[0] = 1u;                                              // count=1, user desc
    g0[1] = lds_byte_off;                                    // lds_addr
    g0[2] = (uint32_t)(ga & 0xFFFFFFFFu);                    // global_addr[31:0]
    g0[3] = (uint32_t)((ga >> 32) & 0x1FFFFFFu) | (2u << 30); // ga[56:32] | type=2
    i32x8 g1;
    // data_size=1 (2B) | pad_enable | pad_interval=3 (16 DW) | pad_amount=3 (4 DW)
    g1[0] = (int)((1u << 16) | (1u << 20) | (3u << 22) | (3u << 25));
    g1[1] = (int)((tensor_d0 & 0xFFFFu) << 16);              // tensor_dim0[15:0]
    g1[2] = (int)((tensor_d0 >> 16) | ((tensor_d1 & 0xFFFFu) << 16));
    g1[3] = (int)((tensor_d1 >> 16) | (tile_d0 << 16));      // | tile_dim0
    g1[4] = (int)(tile_d1 & 0xFFFFu);                        // tile_dim1, tile_dim2=0
    g1[5] = (int)(uint32_t)(stride_d0_elems & 0xFFFFFFFFu);  // dim0_stride[31:0]
    g1[6] = (int)(uint32_t)((stride_d0_elems >> 32) & 0xFFFFu);
    g1[7] = 0;
    i32x4 gz = {0, 0, 0, 0};
#if defined(__clang_major__) && __clang_major__ >= 23
    i32x8 gz8 = {0, 0, 0, 0, 0, 0, 0, 0};
    __builtin_amdgcn_tensor_load_to_lds(g0, g1, gz, gz, gz8, 0);
#else
    __builtin_amdgcn_tensor_load_to_lds(g0, g1, gz, gz, 0);
#endif
}
#else
#define HAVE_TDM 0
#endif

// ---------------------------------------------------------------- f32 -> bf16
__global__ void cvt_f32_to_bf16(const float* __restrict__ in,
                                __bf16* __restrict__ out, int n) {
    int i = blockIdx.x * blockDim.x + threadIdx.x;
    int stride = gridDim.x * blockDim.x;
    for (; i < n; i += stride) out[i] = f2bf(in[i]);
}

// ---------------------------------------------------------------- LayerNorm
__global__ __launch_bounds__(256)
void layernorm_kernel(const float* __restrict__ x, const float* __restrict__ res,
                      const float* __restrict__ gamma, const float* __restrict__ beta,
                      float* __restrict__ outF, __bf16* __restrict__ outB) {
    __shared__ float red_s[8], red_s2[8];
    int row = blockIdx.x;
    const float* xr = x + (size_t)row * D_EMBED;
    const float* rr = res ? res + (size_t)row * D_EMBED : nullptr;
    int t = threadIdx.x;
    float v[4];
    float4 xv = *reinterpret_cast<const float4*>(xr + t * 4);
    v[0] = xv.x; v[1] = xv.y; v[2] = xv.z; v[3] = xv.w;
    if (rr) {
        float4 rv = *reinterpret_cast<const float4*>(rr + t * 4);
        v[0] += rv.x; v[1] += rv.y; v[2] += rv.z; v[3] += rv.w;
    }
    float s = v[0] + v[1] + v[2] + v[3];
    float s2 = v[0]*v[0] + v[1]*v[1] + v[2]*v[2] + v[3]*v[3];
    #pragma unroll
    for (int off = 16; off >= 1; off >>= 1) {
        s  += __shfl_xor(s, off, 32);
        s2 += __shfl_xor(s2, off, 32);
    }
    if ((t & 31) == 0) { red_s[t >> 5] = s; red_s2[t >> 5] = s2; }
    __syncthreads();
    float ts = 0.f, ts2 = 0.f;
    #pragma unroll
    for (int i = 0; i < 8; i++) { ts += red_s[i]; ts2 += red_s2[i]; }
    float mean = ts * (1.0f / D_EMBED);
    float var  = ts2 * (1.0f / D_EMBED) - mean * mean;
    float rstd = rsqrtf(var + 1e-5f);
    #pragma unroll
    for (int i = 0; i < 4; i++) {
        int col = t * 4 + i;
        float y = (v[i] - mean) * rstd * gamma[col] + beta[col];
        size_t idx = (size_t)row * D_EMBED + col;
        if (outF) outF[idx] = y;
        if (outB) outB[idx] = f2bf(y);
    }
}

// ---------------------------------------------------------------- WMMA GEMM
// C[M,N] = epilogue(A[M,K] @ W[N,K]^T + bias).
// Block 256 thr = 8 waves (4 M x 2 N); block tile 128x128; wave tile 32x64
// (2x4 WMMA tiles, 8 WMMA per K-step). A/W K-slabs (128x32) double-buffered
// in LDS via TDM (padded rows, BKP stride), waves read ds_load_b128 fragments.
__global__ __launch_bounds__(256)
void gemm_bf16_kernel(const __bf16* __restrict__ A,  // [M,K]
                      const __bf16* __restrict__ W,  // [N,K]
                      const float* __restrict__ bias,   // [N] or null
                      const float* __restrict__ resid,  // [M,N] or null
                      float* __restrict__ outF,         // [M,N] or null
                      __bf16* __restrict__ outB,        // [M,N] or null
                      int M, int N, int K, int act_silu) {
    __shared__ __bf16 As[2][BM * BKP];   // 2 x 10KB
    __shared__ __bf16 Bs[2][BN * BKP];   // 2 x 10KB
    int lane = threadIdx.x & 31;
    int wave = threadIdx.x >> 5;
    int wm = wave & 3, wn = wave >> 2;
    int g = lane >> 4, lr = lane & 15;
    int m0 = blockIdx.x * BM;
    int n0 = blockIdx.y * BN;

    const f32x8 fz = {0.f,0.f,0.f,0.f,0.f,0.f,0.f,0.f};
    f32x8 c[2][4];
    #pragma unroll
    for (int i = 0; i < 2; i++)
        #pragma unroll
        for (int j = 0; j < 4; j++) c[i][j] = fz;

    int nk = K / BK;
#if HAVE_TDM
    if (wave == 0) {
        tdm_load_2d((uint32_t)(uintptr_t)&As[0][0], A + (size_t)m0 * K,
                    BK, BM, (uint32_t)K, (uint32_t)M, (uint64_t)K);
        tdm_load_2d((uint32_t)(uintptr_t)&Bs[0][0], W + (size_t)n0 * K,
                    BK, BN, (uint32_t)K, (uint32_t)N, (uint64_t)K);
    }
#endif
    for (int ik = 0; ik < nk; ik++) {
        int cur = ik & 1;
#if HAVE_TDM
        if (wave == 0) {
            if (ik + 1 < nk) {
                int nxt = cur ^ 1;
                tdm_load_2d((uint32_t)(uintptr_t)&As[nxt][0],
                            A + (size_t)m0 * K + (size_t)(ik + 1) * BK,
                            BK, BM, (uint32_t)K, (uint32_t)M, (uint64_t)K);
                tdm_load_2d((uint32_t)(uintptr_t)&Bs[nxt][0],
                            W + (size_t)n0 * K + (size_t)(ik + 1) * BK,
                            BK, BN, (uint32_t)K, (uint64_t)N, (uint64_t)K);
                __builtin_amdgcn_s_wait_tensorcnt(2);   // pair ik complete
            } else {
                __builtin_amdgcn_s_wait_tensorcnt(0);
            }
        }
        __syncthreads();
#else
        // cooperative fallback: 256 threads stage both slabs (padded rows)
        {
            int k0 = ik * BK;
            for (int idx = threadIdx.x; idx < BM * 4; idx += 256) {
                int row = idx >> 2, ch = idx & 3;
                *reinterpret_cast<uint4*>(&As[cur][row * BKP + ch * 8]) =
                    *reinterpret_cast<const uint4*>(A + (size_t)(m0 + row) * K + k0 + ch * 8);
                *reinterpret_cast<uint4*>(&Bs[cur][row * BKP + ch * 8]) =
                    *reinterpret_cast<const uint4*>(W + (size_t)(n0 + row) * K + k0 + ch * 8);
            }
        }
        __syncthreads();
#endif
        FragAB a[2], b[4];
        #pragma unroll
        for (int i = 0; i < 2; i++) {
            int row = wm * 32 + i * 16 + lr;
            a[i] = load_fragA(&As[cur][row * BKP], 0, g);
        }
        #pragma unroll
        for (int j = 0; j < 4; j++) {
            int row = wn * 64 + j * 16 + lr;
            b[j] = load_fragB(&Bs[cur][row * BKP], 0, g);
        }
        #pragma unroll
        for (int i = 0; i < 2; i++)
            #pragma unroll
            for (int j = 0; j < 4; j++)
                c[i][j] = wmma_bf16(a[i], b[j], c[i][j]);
        __syncthreads();
    }

    #pragma unroll
    for (int i = 0; i < 2; i++) {
        #pragma unroll
        for (int j = 0; j < 4; j++) {
            int n = n0 + wn * 64 + j * 16 + lr;
            float bn = bias ? bias[n] : 0.f;
            #pragma unroll
            for (int r = 0; r < 8; r++) {
                float vv = c[i][j][r] + bn;
                if (act_silu) vv = vv * (1.0f / (1.0f + __expf(-vv)));
                size_t row = (size_t)(m0 + wm * 32 + i * 16 + r + 8 * g);
                size_t idx = row * (size_t)N + n;
                if (resid) vv += resid[idx];
                if (outF) outF[idx] = vv;
                if (outB) outB[idx] = f2bf(vv);
            }
        }
    }
}

// ---------------------------------------------------------------- Attention
// Flash-attention: 4 independent waves/block, each owns a 16-row q tile of one
// (b,h). Key blocks of 32; scores & PV via WMMA; V^T and P staged in LDS.
__global__ __launch_bounds__(128)
void attention_kernel(const __bf16* __restrict__ Q, const __bf16* __restrict__ Kb,
                      const __bf16* __restrict__ V, float* __restrict__ O) {
    __shared__ __bf16 lds[4][64 * 32 + 16 * 32];   // per wave: Vt[64][32] + Pt[16][32]
    int wave = threadIdx.x >> 5, lane = threadIdx.x & 31;
    int g = lane >> 4, lr = lane & 15;

    int idx = blockIdx.x;
    int qb = idx % (SEQ / 64);
    int h  = (idx / (SEQ / 64)) % N_HEADS;
    int b  = idx / ((SEQ / 64) * N_HEADS);
    int q0 = qb * 64 + wave * 16;

    __bf16* Vt = &lds[wave][0];        // [dim 64][key 32]
    __bf16* Pt = &lds[wave][64 * 32];  // [m 16][k 32]

    const __bf16* qrow = Q + ((size_t)(b * SEQ + q0 + lr)) * D_EMBED + h * HD;
    FragAB qa0 = load_fragA(qrow, 0, g);
    FragAB qa1 = load_fragA(qrow, 32, g);

    const f32x8 fz = {0.f,0.f,0.f,0.f,0.f,0.f,0.f,0.f};
    f32x8 acc[4]; acc[0] = fz; acc[1] = fz; acc[2] = fz; acc[3] = fz;
    float mrow[8], lrow[8];
    #pragma unroll
    for (int r = 0; r < 8; r++) { mrow[r] = -1e30f; lrow[r] = 0.f; }

    for (int kb = 0; kb < SEQ; kb += 32) {
        // stage V^T: this lane handles key (kb + lane)
        const __bf16* vrow = V + ((size_t)(b * SEQ + kb + lane)) * D_EMBED + h * HD;
        #pragma unroll
        for (int cch = 0; cch < 8; cch++) {
            uint4 pk = *reinterpret_cast<const uint4*>(vrow + cch * 8);
            const __bf16* pv = reinterpret_cast<const __bf16*>(&pk);
            #pragma unroll
            for (int e = 0; e < 8; e++) Vt[(cch * 8 + e) * 32 + lane] = pv[e];
        }
        if (kb + 32 < SEQ)
            __builtin_prefetch(V + ((size_t)(b * SEQ + kb + 32 + lane)) * D_EMBED + h * HD, 0, 1);
        // scores: two 16x16 tiles, Kdim = 64 (two WMMAs each)
        f32x8 sc[2];
        #pragma unroll
        for (int tt = 0; tt < 2; tt++) {
            const __bf16* kcol =
                Kb + ((size_t)(b * SEQ + kb + tt * 16 + lr)) * D_EMBED + h * HD;
            FragAB kf0 = load_fragB(kcol, 0, g);
            FragAB kf1 = load_fragB(kcol, 32, g);
            f32x8 z = fz;
            z = wmma_bf16(qa0, kf0, z);
            z = wmma_bf16(qa1, kf1, z);
            sc[tt] = z;
        }
        // online softmax per row (rows m = r + 8g; 16 cols across the 16-lane group)
        #pragma unroll
        for (int r = 0; r < 8; r++) {
            float s0 = sc[0][r] * 0.125f;   // 1/sqrt(64)
            float s1 = sc[1][r] * 0.125f;
            float mx = fmaxf(s0, s1);
            #pragma unroll
            for (int off = 8; off >= 1; off >>= 1) mx = fmaxf(mx, __shfl_xor(mx, off, 16));
            float mnew = fmaxf(mrow[r], mx);
            float corr = __expf(mrow[r] - mnew);
            float p0 = __expf(s0 - mnew), p1 = __expf(s1 - mnew);
            float psum = p0 + p1;
            #pragma unroll
            for (int off = 8; off >= 1; off >>= 1) psum += __shfl_xor(psum, off, 16);
            lrow[r] = lrow[r] * corr + psum;
            mrow[r] = mnew;
            #pragma unroll
            for (int tv = 0; tv < 4; tv++) acc[tv][r] *= corr;
            int m = r + 8 * g;
            Pt[m * 32 + lr]      = f2bf(p0);
            Pt[m * 32 + 16 + lr] = f2bf(p1);
        }
        asm volatile("s_wait_dscnt 0" ::: "memory");  // cross-lane LDS RAW (in-order LDS)
        FragAB pf = load_fragA(Pt + lr * 32, 0, g);   // P as A-fragment (16x32)
        #pragma unroll
        for (int tv = 0; tv < 4; tv++) {
            FragAB vf = load_fragB(Vt + (size_t)(tv * 16 + lr) * 32, 0, g);
            acc[tv] = wmma_bf16(pf, vf, acc[tv]);
        }
    }
    #pragma unroll
    for (int r = 0; r < 8; r++) {
        int m = r + 8 * g;
        float inv = 1.0f / lrow[r];
        #pragma unroll
        for (int tv = 0; tv < 4; tv++) {
            O[((size_t)(b * SEQ + q0 + m)) * D_EMBED + h * HD + tv * 16 + lr] =
                acc[tv][r] * inv;
        }
    }
}

// ---------------------------------------------------------------- launch
extern "C" void kernel_launch(void* const* d_in, const int* in_sizes, int n_in,
                              void* d_out, int out_size, void* d_ws, size_t ws_size,
                              hipStream_t stream) {
    const float* x   = (const float*)d_in[0];
    const float* Wq  = (const float*)d_in[1];
    const float* bq  = (const float*)d_in[2];
    const float* Wk  = (const float*)d_in[3];
    const float* bk  = (const float*)d_in[4];
    const float* Wv  = (const float*)d_in[5];
    const float* bv  = (const float*)d_in[6];
    const float* g1  = (const float*)d_in[7];
    const float* b1  = (const float*)d_in[8];
    const float* g2  = (const float*)d_in[9];
    const float* b2  = (const float*)d_in[10];
    const float* W1  = (const float*)d_in[11];
    const float* bf1 = (const float*)d_in[12];
    const float* W2  = (const float*)d_in[13];
    const float* bf2 = (const float*)d_in[14];
    float* out = (float*)d_out;

    char* wsp = (char*)d_ws;
    auto alloc = [&](size_t bytes) -> void* {
        void* p = (void*)wsp;
        wsp += (bytes + 255) & ~(size_t)255;
        return p;
    };
    __bf16* wq_b = (__bf16*)alloc((size_t)D_EMBED * D_EMBED * 2);
    __bf16* wk_b = (__bf16*)alloc((size_t)D_EMBED * D_EMBED * 2);
    __bf16* wv_b = (__bf16*)alloc((size_t)D_EMBED * D_EMBED * 2);
    __bf16* w1_b = (__bf16*)alloc((size_t)DFF_ * D_EMBED * 2);
    __bf16* w2_b = (__bf16*)alloc((size_t)D_EMBED * DFF_ * 2);
    float*  xn_f = (float*)alloc((size_t)NROWS * D_EMBED * 4);
    __bf16* xn_b = (__bf16*)alloc((size_t)NROWS * D_EMBED * 2);
    __bf16* q_b  = (__bf16*)alloc((size_t)NROWS * D_EMBED * 2);
    __bf16* k_b  = (__bf16*)alloc((size_t)NROWS * D_EMBED * 2);
    __bf16* v_b  = (__bf16*)alloc((size_t)NROWS * D_EMBED * 2);
    float*  at_f = (float*)alloc((size_t)NROWS * D_EMBED * 4);
    __bf16* h_b  = (__bf16*)alloc((size_t)NROWS * D_EMBED * 2);
    __bf16* a1_b = (__bf16*)alloc((size_t)NROWS * DFF_ * 2);
    (void)ws_size; (void)n_in; (void)in_sizes; (void)out_size;

    // 1) weight conversions f32 -> bf16
    int nW = D_EMBED * D_EMBED;
    cvt_f32_to_bf16<<<1024, 256, 0, stream>>>(Wq, wq_b, nW);
    cvt_f32_to_bf16<<<1024, 256, 0, stream>>>(Wk, wk_b, nW);
    cvt_f32_to_bf16<<<1024, 256, 0, stream>>>(Wv, wv_b, nW);
    cvt_f32_to_bf16<<<1024, 256, 0, stream>>>(W1, w1_b, DFF_ * D_EMBED);
    cvt_f32_to_bf16<<<1024, 256, 0, stream>>>(W2, w2_b, D_EMBED * DFF_);

    // 2) LN1: xn = LN(x)
    layernorm_kernel<<<NROWS, 256, 0, stream>>>(x, nullptr, g1, b1, xn_f, xn_b);

    // 3) q/k/v = xn @ W^T + b    (M=4096, N=1024, K=1024)
    dim3 gQ(NROWS / BM, D_EMBED / BN);
    gemm_bf16_kernel<<<gQ, 256, 0, stream>>>(xn_b, wq_b, bq, nullptr, nullptr, q_b,
                                             NROWS, D_EMBED, D_EMBED, 0);
    gemm_bf16_kernel<<<gQ, 256, 0, stream>>>(xn_b, wk_b, bk, nullptr, nullptr, k_b,
                                             NROWS, D_EMBED, D_EMBED, 0);
    gemm_bf16_kernel<<<gQ, 256, 0, stream>>>(xn_b, wv_b, bv, nullptr, nullptr, v_b,
                                             NROWS, D_EMBED, D_EMBED, 0);

    // 4) attention
    attention_kernel<<<BATCH * N_HEADS * (SEQ / 64), 128, 0, stream>>>(q_b, k_b, v_b, at_f);

    // 5) LN2: h = LN(attn + xn)
    layernorm_kernel<<<NROWS, 256, 0, stream>>>(at_f, xn_f, g2, b2, nullptr, h_b);

    // 6) a1 = silu(h @ W1^T + bf1)   (M=4096, N=4096, K=1024)
    dim3 gF1(NROWS / BM, DFF_ / BN);
    gemm_bf16_kernel<<<gF1, 256, 0, stream>>>(h_b, w1_b, bf1, nullptr, nullptr, a1_b,
                                              NROWS, DFF_, D_EMBED, 1);

    // 7) out = a1 @ W2^T + bf2 + xn  (M=4096, N=1024, K=4096)
    dim3 gF2(NROWS / BM, D_EMBED / BN);
    gemm_bf16_kernel<<<gF2, 256, 0, stream>>>(a1_b, w2_b, bf2, xn_f, out, nullptr,
                                              NROWS, D_EMBED, DFF_, 0);
}
